// SelfAttention_29729763623190
// MI455X (gfx1250) — compile-verified, matrix operand order checked by reading, not run on hardware
//
#include <hip/hip_runtime.h>
#include <hip/hip_bf16.h>

typedef __bf16 v16bf __attribute__((ext_vector_type(16)));
typedef float  v8f   __attribute__((ext_vector_type(8)));

#define B_    8
#define C_    384
#define HEAD_ 8
#define CH_   48
#define HW_   16384
#define M3_   1152           // 3*C (stacked wq;wk;wv rows)
#define KP_   392            // LDS K stride (384 + 8 pad, breaks bank alignment)

__device__ __forceinline__ unsigned short f2bf(float f) {
  unsigned int u = __float_as_uint(f);
  u += 0x7FFFu + ((u >> 16) & 1u);          // round-to-nearest-even
  return (unsigned short)(u >> 16);
}
__device__ __forceinline__ float bf2f(unsigned short h) {
  return __uint_as_float(((unsigned int)h) << 16);
}

union Frag {
  v16bf v;
  uint4 u[2];
};

// Load one 16-bf16 WMMA fragment: elems 0-7 = K[klo..klo+7], elems 8-15 = K[klo+16..klo+23]
__device__ __forceinline__ void load_frag(Frag& f, const unsigned short* p) {
  f.u[0] = *(const uint4*)(p);
  f.u[1] = *(const uint4*)(p + 16);
}

// ---------------------------------------------------------------------------
// K0: pack [wq; wk; wv] (f32, 384x384 each) into one 1152x384 bf16 matrix
// ---------------------------------------------------------------------------
__global__ void k0_prep(const float* __restrict__ wq, const float* __restrict__ wk,
                        const float* __restrict__ wv, unsigned short* __restrict__ wqkv) {
  int idx = blockIdx.x * 256 + threadIdx.x;
  if (idx >= 3 * C_ * C_) return;
  const float* src = (idx < C_ * C_) ? wq : ((idx < 2 * C_ * C_) ? wk : wv);
  wqkv[idx] = f2bf(src[idx % (C_ * C_)]);
}

// ---------------------------------------------------------------------------
// K1: Yqkv[b] (1152 x 16384, bf16) = Wqkv (1152x384 bf16) @ X[b] (384x16384 f32)
// Workgroup = one 64-column slab of X staged (transposed, bf16) in LDS,
// reused across all 9 M-blocks. 8 waves, each wave owns 16 M-rows x 64 N.
// ---------------------------------------------------------------------------
__global__ __launch_bounds__(256) void k1_proj(const float* __restrict__ x,
                                               const unsigned short* __restrict__ wqkv,
                                               unsigned short* __restrict__ y) {
  __shared__ unsigned short lds[64 * KP_];            // 50 KB
  const int nb = blockIdx.x;                          // 256 n-blocks of 64
  const int b  = blockIdx.y;
  const int tid = threadIdx.x;

  const float* xb = x + ((size_t)b * C_) * HW_ + (size_t)nb * 64;
  for (int idx = tid; idx < C_ * 64; idx += 256) {
    int k = idx >> 6, n = idx & 63;                   // coalesced along n
    lds[n * KP_ + k] = f2bf(xb[(size_t)k * HW_ + n]); // LDS layout [n][k]
  }
  __syncthreads();

  const int wave = tid >> 5, lane = tid & 31;
  const int l16 = lane & 15, lhalf = lane >> 4;
  unsigned short* yb = y + ((size_t)b * M3_) * HW_ + (size_t)nb * 64;

  for (int mb = 0; mb < 9; ++mb) {
    const int mrow = mb * 128 + wave * 16 + l16;
    const unsigned short* pa = wqkv + (size_t)mrow * C_;
    __builtin_prefetch(pa, 0, 1);
    v8f acc0 = {}, acc1 = {}, acc2 = {}, acc3 = {};
    for (int ks = 0; ks < 12; ++ks) {
      const int klo = ks * 32 + lhalf * 8;
      Frag a; load_frag(a, pa + klo);
      Frag bb;
      load_frag(bb, &lds[(0 * 16 + l16) * KP_ + klo]);
      acc0 = __builtin_amdgcn_wmma_f32_16x16x32_bf16(false, a.v, false, bb.v, (short)0, acc0, false, false);
      load_frag(bb, &lds[(1 * 16 + l16) * KP_ + klo]);
      acc1 = __builtin_amdgcn_wmma_f32_16x16x32_bf16(false, a.v, false, bb.v, (short)0, acc1, false, false);
      load_frag(bb, &lds[(2 * 16 + l16) * KP_ + klo]);
      acc2 = __builtin_amdgcn_wmma_f32_16x16x32_bf16(false, a.v, false, bb.v, (short)0, acc2, false, false);
      load_frag(bb, &lds[(3 * 16 + l16) * KP_ + klo]);
      acc3 = __builtin_amdgcn_wmma_f32_16x16x32_bf16(false, a.v, false, bb.v, (short)0, acc3, false, false);
    }
    const int rbase = mb * 128 + wave * 16 + (lhalf ? 8 : 0);
#pragma unroll
    for (int v = 0; v < 8; ++v) {
      const size_t ro = (size_t)(rbase + v) * HW_;
      yb[ro +  0 + l16] = f2bf(acc0[v]);
      yb[ro + 16 + l16] = f2bf(acc1[v]);
      yb[ro + 32 + l16] = f2bf(acc2[v]);
      yb[ro + 48 + l16] = f2bf(acc3[v]);
    }
  }
}

// ---------------------------------------------------------------------------
// K2: per-row reciprocal L2 norms of q and k (rows over hw). 1 block per row.
// ---------------------------------------------------------------------------
__global__ __launch_bounds__(256) void k2_norms(const unsigned short* __restrict__ y,
                                                float* __restrict__ rq,
                                                float* __restrict__ rk) {
  __shared__ float red[256];
  const int bid = blockIdx.x;
  const int which = bid >= (B_ * C_);
  const int r = which ? bid - B_ * C_ : bid;
  const int b = r / C_, cix = r % C_;
  const unsigned short* p = y + ((size_t)b * M3_ + (which ? C_ : 0) + cix) * HW_;
  const uint4* p4 = (const uint4*)p;
  float s = 0.f;
  for (int i = threadIdx.x; i < HW_ / 8; i += 256) {
    uint4 u = p4[i];
    unsigned int ws[4] = {u.x, u.y, u.z, u.w};
#pragma unroll
    for (int q = 0; q < 4; ++q) {
      float a = __uint_as_float((ws[q] & 0xFFFFu) << 16);
      float c = __uint_as_float(ws[q] & 0xFFFF0000u);
      s += a * a + c * c;
    }
  }
  red[threadIdx.x] = s;
  __syncthreads();
  for (int off = 128; off > 0; off >>= 1) {
    if (threadIdx.x < off) red[threadIdx.x] += red[threadIdx.x + off];
    __syncthreads();
  }
  if (threadIdx.x == 0)
    (which ? rk : rq)[r] = 1.f / fmaxf(sqrtf(red[0]), 1e-12f);
}

// ---------------------------------------------------------------------------
// K3: per (b,head) scores S = q k^T (48x48, K=16384 via WMMA bf16),
// scaled by 1/(||q_i|| ||k_j||) * temperature, then row softmax.
// 8 waves split K; reduction via LDS float atomics (ds_add_f32).
// ---------------------------------------------------------------------------
__global__ __launch_bounds__(256) void k3_attn(const unsigned short* __restrict__ y,
                                               const float* __restrict__ rq,
                                               const float* __restrict__ rk,
                                               const float* __restrict__ temp,
                                               float* __restrict__ attn) {
  __shared__ float ssum[CH_ * CH_];
  const int b = blockIdx.x >> 3, h = blockIdx.x & 7;
  const int tid = threadIdx.x, wave = tid >> 5, lane = tid & 31;
  const int l16 = lane & 15, lhalf = lane >> 4;

  for (int i = tid; i < CH_ * CH_; i += 256) ssum[i] = 0.f;
  __syncthreads();

  const unsigned short* qb = y + ((size_t)b * M3_ + h * CH_) * HW_;
  const unsigned short* kb = y + ((size_t)b * M3_ + C_ + h * CH_) * HW_;
  const int n0 = wave * 2048;

  v8f acc[3][3] = {};
  for (int ks = 0; ks < 64; ++ks) {
    const int klo = n0 + ks * 32 + lhalf * 8;
    Frag aq[3], bk[3];
#pragma unroll
    for (int t = 0; t < 3; ++t) {
      load_frag(aq[t], qb + (size_t)(t * 16 + l16) * HW_ + klo);
      load_frag(bk[t], kb + (size_t)(t * 16 + l16) * HW_ + klo);
    }
#pragma unroll
    for (int i = 0; i < 3; ++i)
#pragma unroll
      for (int j = 0; j < 3; ++j)
        acc[i][j] = __builtin_amdgcn_wmma_f32_16x16x32_bf16(
            false, aq[i].v, false, bk[j].v, (short)0, acc[i][j], false, false);
  }
#pragma unroll
  for (int i = 0; i < 3; ++i)
#pragma unroll
    for (int j = 0; j < 3; ++j)
#pragma unroll
      for (int v = 0; v < 8; ++v) {
        int row = i * 16 + (lhalf ? v + 8 : v);
        int col = j * 16 + l16;
        atomicAdd(&ssum[row * CH_ + col], acc[i][j][v]);
      }
  __syncthreads();

  if (tid < CH_) {
    const int i = tid;
    const float ri = rq[b * C_ + h * CH_ + i];
    const float tv = temp[h];
    float vals[CH_];
    float m = -1e30f;
#pragma unroll
    for (int j = 0; j < CH_; ++j) {
      vals[j] = ssum[i * CH_ + j] * ri * rk[b * C_ + h * CH_ + j] * tv;
      m = fmaxf(m, vals[j]);
    }
    float s = 0.f;
#pragma unroll
    for (int j = 0; j < CH_; ++j) { vals[j] = __expf(vals[j] - m); s += vals[j]; }
    const float inv = 1.f / s;
    float* arow = attn + (((size_t)b * HEAD_ + h) * CH_ + i) * CH_;
#pragma unroll
    for (int j = 0; j < CH_; ++j) arow[j] = vals[j] * inv;
  }
}

// ---------------------------------------------------------------------------
// K4: W2[b] = wo @ blockdiag(attn[b])  (fuses attn@v projection into wo GEMM)
// W2[b][o][h*48+d] = sum_c wo[o][h*48+c] * attn[b,h,c,d]   -> bf16
// ---------------------------------------------------------------------------
__global__ __launch_bounds__(256) void k4_w2(const float* __restrict__ attn,
                                             const float* __restrict__ wo,
                                             unsigned short* __restrict__ w2) {
  __shared__ float sa[CH_ * CH_];
  const int b = blockIdx.x >> 3, h = blockIdx.x & 7;
  const float* ab = attn + ((size_t)b * HEAD_ + h) * CH_ * CH_;
  for (int i = threadIdx.x; i < CH_ * CH_; i += 256) sa[i] = ab[i];
  __syncthreads();
  for (int pr = threadIdx.x; pr < C_ * CH_; pr += 256) {
    const int o = pr / CH_, d = pr % CH_;
    const float* wrow = wo + (size_t)o * C_ + h * CH_;
    float s = 0.f;
#pragma unroll
    for (int c = 0; c < CH_; ++c) s += wrow[c] * sa[c * CH_ + d];
    w2[((size_t)b * C_ + o) * C_ + h * CH_ + d] = f2bf(s);
  }
}

// ---------------------------------------------------------------------------
// K5: out[b] (384x16384 f32) = W2[b] (384x384 bf16) @ v[b] (384x16384 bf16)
// Same structure as K1; v slab staged in LDS (already bf16).
// ---------------------------------------------------------------------------
__global__ __launch_bounds__(256) void k5_out(const unsigned short* __restrict__ y,
                                              const unsigned short* __restrict__ w2,
                                              float* __restrict__ out) {
  __shared__ unsigned short lds[64 * KP_];
  const int nb = blockIdx.x;
  const int b  = blockIdx.y;
  const int tid = threadIdx.x;

  const unsigned short* vb = y + ((size_t)b * M3_ + 2 * C_) * HW_ + (size_t)nb * 64;
  for (int idx = tid; idx < C_ * 64; idx += 256) {
    int k = idx >> 6, n = idx & 63;
    lds[n * KP_ + k] = vb[(size_t)k * HW_ + n];
  }
  __syncthreads();

  const int wave = tid >> 5, lane = tid & 31;
  const int l16 = lane & 15, lhalf = lane >> 4;
  const unsigned short* w2b = w2 + (size_t)b * C_ * C_;
  float* ob = out + ((size_t)b * C_) * HW_ + (size_t)nb * 64;

  for (int mb = 0; mb < 3; ++mb) {
    const int mrow = mb * 128 + wave * 16 + l16;
    const unsigned short* pa = w2b + (size_t)mrow * C_;
    __builtin_prefetch(pa, 0, 1);
    v8f acc0 = {}, acc1 = {}, acc2 = {}, acc3 = {};
    for (int ks = 0; ks < 12; ++ks) {
      const int klo = ks * 32 + lhalf * 8;
      Frag a; load_frag(a, pa + klo);
      Frag bb;
      load_frag(bb, &lds[(0 * 16 + l16) * KP_ + klo]);
      acc0 = __builtin_amdgcn_wmma_f32_16x16x32_bf16(false, a.v, false, bb.v, (short)0, acc0, false, false);
      load_frag(bb, &lds[(1 * 16 + l16) * KP_ + klo]);
      acc1 = __builtin_amdgcn_wmma_f32_16x16x32_bf16(false, a.v, false, bb.v, (short)0, acc1, false, false);
      load_frag(bb, &lds[(2 * 16 + l16) * KP_ + klo]);
      acc2 = __builtin_amdgcn_wmma_f32_16x16x32_bf16(false, a.v, false, bb.v, (short)0, acc2, false, false);
      load_frag(bb, &lds[(3 * 16 + l16) * KP_ + klo]);
      acc3 = __builtin_amdgcn_wmma_f32_16x16x32_bf16(false, a.v, false, bb.v, (short)0, acc3, false, false);
    }
    const int rbase = mb * 128 + wave * 16 + (lhalf ? 8 : 0);
#pragma unroll
    for (int v = 0; v < 8; ++v) {
      const size_t ro = (size_t)(rbase + v) * HW_;
      ob[ro +  0 + l16] = acc0[v];
      ob[ro + 16 + l16] = acc1[v];
      ob[ro + 32 + l16] = acc2[v];
      ob[ro + 48 + l16] = acc3[v];
    }
  }
}

// ---------------------------------------------------------------------------
extern "C" void kernel_launch(void* const* d_in, const int* in_sizes, int n_in,
                              void* d_out, int out_size, void* d_ws, size_t ws_size,
                              hipStream_t stream) {
  const float* x    = (const float*)d_in[0];
  const float* wq   = (const float*)d_in[1];
  const float* wk   = (const float*)d_in[2];
  const float* wv   = (const float*)d_in[3];
  const float* wo   = (const float*)d_in[4];
  const float* temp = (const float*)d_in[5];

  char* ws = (char*)d_ws;
  unsigned short* wqkv = (unsigned short*)ws; ws += (size_t)3 * C_ * C_ * 2;
  unsigned short* y    = (unsigned short*)ws; ws += (size_t)B_ * M3_ * HW_ * 2;
  float*          rq   = (float*)ws;          ws += (size_t)B_ * C_ * 4;
  float*          rk   = (float*)ws;          ws += (size_t)B_ * C_ * 4;
  float*          attn = (float*)ws;          ws += (size_t)B_ * HEAD_ * CH_ * CH_ * 4;
  unsigned short* w2   = (unsigned short*)ws; ws += (size_t)B_ * C_ * C_ * 2;
  float* out = (float*)d_out;

  k0_prep<<<(3 * C_ * C_ + 255) / 256, 256, 0, stream>>>(wq, wk, wv, wqkv);
  k1_proj<<<dim3(HW_ / 64, B_), 256, 0, stream>>>(x, wqkv, y);
  k2_norms<<<2 * B_ * C_, 256, 0, stream>>>(y, rq, rk);
  k3_attn<<<B_ * HEAD_, 256, 0, stream>>>(y, rq, rk, temp, attn);
  k4_w2<<<B_ * HEAD_, 256, 0, stream>>>(attn, wo, w2);
  k5_out<<<dim3(HW_ / 64, B_), 256, 0, stream>>>(y, w2, out);
}
